// MultiheadAttention_51101520888222
// MI455X (gfx1250) — matrix operand order repaired and checked by reading, not compiled, not run on hardware
//
#include <hip/hip_runtime.h>
#include <hip/hip_bf16.h>
#include <cstdint>
#include <cstddef>

// Problem constants (match reference)
#define B_ 2
#define S_ 2048
#define E_ 768
#define H_ 12
#define D_ 64

typedef __attribute__((ext_vector_type(16))) __bf16 v16bf;
typedef __attribute__((ext_vector_type(8)))  float  v8f;

union Frag16 { v16bf v; unsigned short h[16]; unsigned int w[8]; };
union FragC  { v8f   v; float f[8]; };

// f32 -> bf16, round-to-nearest-even (cold paths only)
__device__ __forceinline__ unsigned short f2bf(float x) {
  unsigned u = __float_as_uint(x);
  u += 0x7FFFu + ((u >> 16) & 1u);
  return (unsigned short)(u >> 16);
}
__device__ __forceinline__ unsigned int pack_bf2(float a, float b) {
  return (unsigned int)f2bf(a) | ((unsigned int)f2bf(b) << 16);
}

// A-fragment (16x32 bf16, MxK): lane holds row M=lane%16.
// VGPR j (pair of K): K = 2j (j<4) else 2j+8; lanes>=16 add +8.
__device__ __forceinline__ int koffA(int j, int hi) {
  return 2 * j + ((j < 4) ? 0 : 8) + (hi ? 8 : 0);
}
// B-fragment (32x16 bf16, KxN): lane holds col N=lane%16.
// VGPR j (pair of K): K = 2j (j<4) else 2j+8; lanes>=16 add +8.
__device__ __forceinline__ int koffB(int j, int hi) {
  return 2 * j + ((j < 4) ? 0 : 8) + (hi ? 8 : 0);
}

// ---------------------------------------------------------------------------
// Kernel 0: bulk f32 -> bf16 conversion (8 elements/thread, B128 in/out).
// Streams once; results live in the 192MB L2 for the GEMM kernels.
// ---------------------------------------------------------------------------
__global__ __launch_bounds__(256)
void cvt_bf16_kernel(const float* __restrict__ src,
                     unsigned int* __restrict__ dst, int n8) {
  const int i = blockIdx.x * blockDim.x + threadIdx.x;
  if (i >= n8) return;
  const float4* s4 = (const float4*)src;
  const float4 a = s4[2 * i], b = s4[2 * i + 1];
  uint4 o;
  o.x = pack_bf2(a.x, a.y);
  o.y = pack_bf2(a.z, a.w);
  o.z = pack_bf2(b.x, b.y);
  o.w = pack_bf2(b.z, b.w);
  ((uint4*)dst)[i] = o;
}

// ---------------------------------------------------------------------------
// Kernel 1: fused QKV projection.  y = x @ W^T  (x, W pre-converted to bf16)
// grid = (M/16, E/16, 3)  block = 32 (one wave computes a 16x16 tile)
// Inner loop: 16 dword loads + 1 WMMA (no conversion VALU).
// z=0 -> Q stored [B,H,S,D] ; z=1 -> K stored [B,H,S,D]
// z=2 -> V stored transposed [B,H,D,S] (so PV B-fragments are dword loads)
// ---------------------------------------------------------------------------
__global__ __launch_bounds__(32)
void proj_kernel(const unsigned short* __restrict__ xq,
                 const unsigned short* __restrict__ xk,
                 const unsigned short* __restrict__ xv,
                 const unsigned short* __restrict__ wq,
                 const unsigned short* __restrict__ wk,
                 const unsigned short* __restrict__ wv,
                 unsigned short* __restrict__ qb,
                 unsigned short* __restrict__ kb,
                 unsigned short* __restrict__ vtb) {
  const int lane = threadIdx.x;
  const int hi = lane >> 4;
  const int lr = lane & 15;
  const int mBase = blockIdx.x * 16;
  const int oBase = blockIdx.y * 16;
  const int which = blockIdx.z;

  const unsigned short* x = (which == 0) ? xq : ((which == 1) ? xk : xv);
  const unsigned short* W = (which == 0) ? wq : ((which == 1) ? wk : wv);

  const unsigned short* xrow = x + (size_t)(mBase + lr) * E_;
  const unsigned short* wrow = W + (size_t)(oBase + lr) * E_;

  FragC c;
#pragma unroll
  for (int i = 0; i < 8; ++i) c.f[i] = 0.0f;

  for (int k0 = 0; k0 < E_; k0 += 32) {
    Frag16 a, b;
#pragma unroll
    for (int j = 0; j < 8; ++j) {
      a.w[j] = *(const unsigned int*)(xrow + k0 + koffA(j, hi));
      b.w[j] = *(const unsigned int*)(wrow + k0 + koffB(j, hi));
    }
    c.v = __builtin_amdgcn_wmma_f32_16x16x32_bf16(false, a.v, false, b.v,
                                                  (short)0, c.v, false, false);
  }

  // Epilogue: tile never crosses a batch boundary (mBase % 16 == 0, S=2048).
  const int o = oBase + lr;              // output feature
  const int h = o >> 6;                  // head
  const int d = o & 63;                  // within-head dim
  const int m0 = mBase + hi * 8;         // first row this half-wave holds
  const int bb = m0 >> 11;
  const int s0 = m0 & (S_ - 1);

  if (which == 2) {
    // transposed V: r-stride is 1 -> pack pairs, dword stores
    unsigned short* dst = vtb + (((size_t)bb * H_ + h) * D_ + d) * S_ + s0;
#pragma unroll
    for (int r = 0; r < 8; r += 2)
      *(unsigned int*)(dst + r) = pack_bf2(c.f[r], c.f[r + 1]);
  } else {
    unsigned short* dst = ((which == 0) ? qb : kb) +
                          (((size_t)bb * H_ + h) * S_ + s0) * D_ + d;
#pragma unroll
    for (int r = 0; r < 8; ++r) dst[(size_t)r * D_] = f2bf(c.f[r]);
  }
}

// ---------------------------------------------------------------------------
// Kernel 2: scores = (Q K^T)/sqrt(D) with key-padding mask, raw f32 -> attn.
// grid = (S/16, H, B), block = 32. One wave: 16 query rows x all 2048 keys.
// K tiles (256KB per head) are shared by 128 blocks -> L2 resident.
// ---------------------------------------------------------------------------
__global__ __launch_bounds__(32)
void scores_kernel(const unsigned short* __restrict__ qb,
                   const unsigned short* __restrict__ kb,
                   const int* __restrict__ mask,
                   float* __restrict__ attn) {
  const int lane = threadIdx.x;
  const int hi = lane >> 4;
  const int lr = lane & 15;
  const int mBase = blockIdx.x * 16;
  const int h = blockIdx.y, b = blockIdx.z;
  const size_t bh = (size_t)b * H_ + h;

  // Load both Q A-fragments (d=0..31 and d=32..63) once.
  const unsigned short* qrow = qb + (bh * S_ + mBase + lr) * D_;
  Frag16 a0, a1;
#pragma unroll
  for (int j = 0; j < 8; ++j) {
    const int ka = koffA(j, hi);
    a0.w[j] = *(const unsigned int*)(qrow + ka);
    a1.w[j] = *(const unsigned int*)(qrow + 32 + ka);
  }

  const int* mrow = mask + b * S_;
  const unsigned short* kbh = kb + bh * S_ * D_;
  float* abase = attn + bh * (size_t)S_ * S_;

  for (int t0 = 0; t0 < S_; t0 += 16) {
    const unsigned short* krow = kbh + (size_t)(t0 + lr) * D_;
    // prefetch next K tile row (lowers to global_prefetch_b8)
    if (t0 + 16 < S_) __builtin_prefetch(krow + 16 * D_, 0, 3);
    Frag16 b0, b1;
#pragma unroll
    for (int j = 0; j < 8; ++j) {
      const int kd = koffB(j, hi);
      b0.w[j] = *(const unsigned int*)(krow + kd);
      b1.w[j] = *(const unsigned int*)(krow + 32 + kd);
    }
    FragC c;
#pragma unroll
    for (int i = 0; i < 8; ++i) c.f[i] = 0.0f;
    c.v = __builtin_amdgcn_wmma_f32_16x16x32_bf16(false, a0.v, false, b0.v,
                                                  (short)0, c.v, false, false);
    c.v = __builtin_amdgcn_wmma_f32_16x16x32_bf16(false, a1.v, false, b1.v,
                                                  (short)0, c.v, false, false);
    const int t = t0 + lr;                 // C frag: N = lane%16
    const bool pad = (mrow[t] != 0);
#pragma unroll
    for (int r = 0; r < 8; ++r) {          // C frag: M = r + 8*hi
      const int srow = mBase + r + hi * 8;
      const float val = pad ? -__builtin_inff() : c.f[r] * 0.125f;
      abase[(size_t)srow * S_ + t] = val;
    }
  }
}

// ---------------------------------------------------------------------------
// Kernel 3: per-row softmax stats (max, 1/sum(exp)). One wave per row,
// float4 (B128) streaming loads.
// grid = B*H*S/8, block = 256 (8 waves).
// ---------------------------------------------------------------------------
__global__ __launch_bounds__(256)
void softmax_stats_kernel(const float* __restrict__ attn,
                          float2* __restrict__ stats) {
  const int wave = threadIdx.x >> 5;
  const int lane = threadIdx.x & 31;
  const size_t row = (size_t)blockIdx.x * 8 + wave;
  const float4* p4 = (const float4*)(attn + row * S_);

  float m = -__builtin_inff();
#pragma unroll 4
  for (int i = lane; i < S_ / 4; i += 32) {
    const float4 v = p4[i];
    m = fmaxf(m, fmaxf(fmaxf(v.x, v.y), fmaxf(v.z, v.w)));
  }
#pragma unroll
  for (int off = 16; off; off >>= 1) m = fmaxf(m, __shfl_xor(m, off));

  float s = 0.0f;
#pragma unroll 4
  for (int i = lane; i < S_ / 4; i += 32) {
    const float4 v = p4[i];
    s += __expf(v.x - m) + __expf(v.y - m) + __expf(v.z - m) + __expf(v.w - m);
  }
#pragma unroll
  for (int off = 16; off; off >>= 1) s += __shfl_xor(s, off);

  if (lane == 0) stats[row] = make_float2(m, 1.0f / s);
}

// ---------------------------------------------------------------------------
// Kernel 4: normalize probs in place (this IS the reference `attn` output)
// and accumulate out[16x64] = P @ V with 4 WMMAs per 32-key chunk.
// grid = (S/16, H, B), block = 32.
// ---------------------------------------------------------------------------
__global__ __launch_bounds__(32)
void pv_kernel(float* __restrict__ attn,
               const unsigned short* __restrict__ vtb,  // [B,H,D,S] bf16
               const float2* __restrict__ stats,
               float* __restrict__ out) {               // [B,S,E]
  const int lane = threadIdx.x;
  const int hi = lane >> 4;
  const int lr = lane & 15;
  const int mBase = blockIdx.x * 16;
  const int h = blockIdx.y, b = blockIdx.z;
  const size_t bh = (size_t)b * H_ + h;

  const size_t rowg = bh * S_ + mBase + lr;   // this lane's A-frag row
  const float2 st = stats[rowg];
  const float m = st.x, invl = st.y;
  float* arow = attn + rowg * (size_t)S_;

  FragC c0, c1, c2, c3;
#pragma unroll
  for (int i = 0; i < 8; ++i) { c0.f[i] = c1.f[i] = c2.f[i] = c3.f[i] = 0.0f; }

  const unsigned short* vbase = vtb + (bh * D_) * (size_t)S_;

  for (int t0 = 0; t0 < S_; t0 += 32) {
    if (t0 + 32 < S_) __builtin_prefetch(arow + t0 + 32, 0, 3);
    Frag16 a;
#pragma unroll
    for (int j = 0; j < 8; ++j) {
      const int ta = t0 + koffA(j, hi);
      const float2 sv = *(const float2*)(arow + ta);
      const float p0 = __expf(sv.x - m) * invl;
      const float p1 = __expf(sv.y - m) * invl;
      *(float2*)(arow + ta) = make_float2(p0, p1);   // normalized attn out
      a.w[j] = pack_bf2(p0, p1);
    }
    Frag16 b0, b1, b2, b3;
#pragma unroll
    for (int j = 0; j < 8; ++j) {
      const int tb = t0 + koffB(j, hi);
      b0.w[j] = *(const unsigned int*)(vbase + (size_t)(lr +  0) * S_ + tb);
      b1.w[j] = *(const unsigned int*)(vbase + (size_t)(lr + 16) * S_ + tb);
      b2.w[j] = *(const unsigned int*)(vbase + (size_t)(lr + 32) * S_ + tb);
      b3.w[j] = *(const unsigned int*)(vbase + (size_t)(lr + 48) * S_ + tb);
    }
    c0.v = __builtin_amdgcn_wmma_f32_16x16x32_bf16(false, a.v, false, b0.v,
                                                   (short)0, c0.v, false, false);
    c1.v = __builtin_amdgcn_wmma_f32_16x16x32_bf16(false, a.v, false, b1.v,
                                                   (short)0, c1.v, false, false);
    c2.v = __builtin_amdgcn_wmma_f32_16x16x32_bf16(false, a.v, false, b2.v,
                                                   (short)0, c2.v, false, false);
    c3.v = __builtin_amdgcn_wmma_f32_16x16x32_bf16(false, a.v, false, b3.v,
                                                   (short)0, c3.v, false, false);
  }

#pragma unroll
  for (int r = 0; r < 8; ++r) {
    const int s = mBase + r + hi * 8;
    const size_t obase = ((size_t)b * S_ + s) * E_ + (size_t)h * 64;
    out[obase +  0 + lr] = c0.f[r];
    out[obase + 16 + lr] = c1.f[r];
    out[obase + 32 + lr] = c2.f[r];
    out[obase + 48 + lr] = c3.f[r];
  }
}

// ---------------------------------------------------------------------------
extern "C" void kernel_launch(void* const* d_in, const int* in_sizes, int n_in,
                              void* d_out, int out_size, void* d_ws,
                              size_t ws_size, hipStream_t stream) {
  (void)in_sizes; (void)n_in; (void)out_size; (void)ws_size;

  const float* q_in = (const float*)d_in[0];
  const float* k_in = (const float*)d_in[1];
  const float* v_in = (const float*)d_in[2];
  const int*   mask = (const int*)d_in[3];
  const float* Wq = (const float*)d_in[4];
  const float* Wk = (const float*)d_in[5];
  const float* Wv = (const float*)d_in[6];

  float* outp = (float*)d_out;                         // [B,S,E]
  float* attn = outp + (size_t)B_ * S_ * E_;           // [B,H,S,S]

  // Workspace layout (bf16 buffers as ushort)
  const size_t n_proj = (size_t)B_ * H_ * S_ * D_;     // 3,145,728 per buffer
  const size_t n_x    = (size_t)B_ * S_ * E_;          // 3,145,728 per input
  const size_t n_w    = (size_t)E_ * E_;               // 589,824 per weight
  unsigned short* qb   = (unsigned short*)d_ws;
  unsigned short* kb   = qb + n_proj;
  unsigned short* vtb  = kb + n_proj;
  unsigned short* xq   = vtb + n_proj;
  unsigned short* xk   = xq + n_x;
  unsigned short* xv   = xk + n_x;
  unsigned short* wqb  = xv + n_x;
  unsigned short* wkb  = wqb + n_w;
  unsigned short* wvb  = wkb + n_w;
  float2* stats = (float2*)(wvb + n_w);                // [B*H*S]

  // Pre-convert inputs/weights to bf16 (streaming, L2-resident afterwards).
  const int nx8 = (int)(n_x / 8), nw8 = (int)(n_w / 8);
  cvt_bf16_kernel<<<(nx8 + 255) / 256, 256, 0, stream>>>(q_in, (unsigned int*)xq, nx8);
  cvt_bf16_kernel<<<(nx8 + 255) / 256, 256, 0, stream>>>(k_in, (unsigned int*)xk, nx8);
  cvt_bf16_kernel<<<(nx8 + 255) / 256, 256, 0, stream>>>(v_in, (unsigned int*)xv, nx8);
  cvt_bf16_kernel<<<(nw8 + 255) / 256, 256, 0, stream>>>(Wq, (unsigned int*)wqb, nw8);
  cvt_bf16_kernel<<<(nw8 + 255) / 256, 256, 0, stream>>>(Wk, (unsigned int*)wkb, nw8);
  cvt_bf16_kernel<<<(nw8 + 255) / 256, 256, 0, stream>>>(Wv, (unsigned int*)wvb, nw8);

  dim3 g1(B_ * S_ / 16, E_ / 16, 3);
  proj_kernel<<<g1, 32, 0, stream>>>(xq, xk, xv, wqb, wkb, wvb, qb, kb, vtb);

  dim3 g2(S_ / 16, H_, B_);
  scores_kernel<<<g2, 32, 0, stream>>>(qb, kb, mask, attn);

  softmax_stats_kernel<<<(B_ * H_ * S_) / 8, 256, 0, stream>>>(attn, stats);

  pv_kernel<<<g2, 32, 0, stream>>>(attn, vtb, stats, outp);
}
